// DOMTransformer_8323646619771
// MI455X (gfx1250) — matrix operand order repaired and checked by reading, not compiled, not run on hardware
//
#include <hip/hip_runtime.h>

// ---------------------------------------------------------------------------
// CDNA5 (gfx1250) DOM-transformer. wave32, WMMA bf16 (f32 accumulate),
// async global->LDS staging for GEMM operand tiles.
// ---------------------------------------------------------------------------

typedef __attribute__((ext_vector_type(16))) __bf16 v16bf;
typedef __attribute__((ext_vector_type(8)))  __bf16 v8bf;
typedef __attribute__((ext_vector_type(8)))  float  v8f;
typedef int v4i __attribute__((vector_size(16)));   // matches builtin prototype

// problem constants
static constexpr int BB   = 64;
static constexpr int SS   = 512;
static constexpr int PF   = 4;
static constexpr int DD   = 128;
static constexpr int NHh  = 8;
static constexpr int LL   = 4;
static constexpr int DFFq = 512;
static constexpr int TDq  = 2048;
static constexpr int MT   = BB * SS;      // 32768 tokens

// ---------------------------------------------------------------------------
// CDNA5 async global->LDS copy (ASYNCcnt path), with portable fallback
// ---------------------------------------------------------------------------
#if defined(__HIP_DEVICE_COMPILE__) &&                                         \
    __has_builtin(__builtin_amdgcn_global_load_async_to_lds_b128) &&           \
    __has_builtin(__builtin_amdgcn_s_wait_asynccnt)
#define USE_ASYNC_COPY 1
#else
#define USE_ASYNC_COPY 0
#endif

__device__ __forceinline__ void async_copy16(const __bf16* g, __bf16* l)
{
#if USE_ASYNC_COPY
  v4i* gg = (v4i*)g;   // strip const, re-pun in generic AS
  v4i* ll = (v4i*)l;
  __builtin_amdgcn_global_load_async_to_lds_b128(
      (__attribute__((address_space(1))) v4i*)gg,
      (__attribute__((address_space(3))) v4i*)ll, 0, 0);
#else
  *(v8bf*)l = *(const v8bf*)g;
#endif
}

__device__ __forceinline__ void async_wait_all()
{
#if USE_ASYNC_COPY
  __builtin_amdgcn_s_wait_asynccnt(0);
#endif
}

// ---------------------------------------------------------------------------
// weight transpose + f32->bf16:  src [Kd][Nd] (row major) -> dst [Nd][Kd]
// ---------------------------------------------------------------------------
__global__ void k_wt(const float* __restrict__ src, __bf16* __restrict__ dst,
                     int Kd, int Nd)
{
  int i = blockIdx.x * 256 + threadIdx.x;
  if (i >= Kd * Nd) return;
  int k = i / Nd, n = i - k * Nd;
  dst[(size_t)n * Kd + k] = (__bf16)src[i];
}

// f32 -> bf16 elementwise
__global__ void k_cvt(const float* __restrict__ src, __bf16* __restrict__ dst, int n)
{
  int i = blockIdx.x * 256 + threadIdx.x;
  if (i < n) dst[i] = (__bf16)src[i];
}

// h = packed @ Wp + bp   (PF=4 -> scalar FMA is fine)
__global__ void k_embed(const float* __restrict__ x, const float* __restrict__ Wp,
                        const float* __restrict__ bp, float* __restrict__ h, int n)
{
  int i = blockIdx.x * 256 + threadIdx.x;
  if (i >= n) return;
  int m = i >> 7, d = i & 127;
  float acc = bp[d];
#pragma unroll
  for (int p = 0; p < PF; ++p) acc += x[m * PF + p] * Wp[p * DD + d];
  h[i] = acc;
}

// ---------------------------------------------------------------------------
// WMMA GEMM: C[M,N] = A[M,K](bf16) @ Wt[N,K](bf16)^T + bias, opt exact GELU,
// output f32 or bf16.  Block = 256 thr = 8 waves, tile 32(M) x 64(N).
// Per 32-wide k-chunk both operand tiles are DMA'd into LDS via the async
// global->LDS path (ASYNCcnt), then WMMA fragments come from ds_load_b128.
// LDS row stride = 40 elems (80B = 20 banks): 16 consecutive rows hit 16
// distinct bank groups -> conflict-free b128 fragment loads.
// A fragment (16-bit A 16x32): lane -> row (lane&15); lo = K[hlf*8..+7],
// hi = K[16+hlf*8..+7].  B fragment (16-bit B 32x16): lane -> col (lane&15);
// lo = K[hlf*8..+7], hi = K[16+hlf*8..+7] (same k order as A).
// ---------------------------------------------------------------------------
template <int ACT, int OUTB>
__global__ __launch_bounds__(256)
void k_gemm(const __bf16* __restrict__ A, const __bf16* __restrict__ Wt,
            const float* __restrict__ bias, float* __restrict__ Cf,
            __bf16* __restrict__ Cb, int M, int N, int K)
{
  constexpr int LDA = 40;                       // padded LDS row stride (elems)
  __shared__ __align__(16) __bf16 sA[32 * LDA];
  __shared__ __align__(16) __bf16 sB[64 * LDA];

  const int tid  = threadIdx.x;
  const int wave = tid >> 5;
  const int lane = tid & 31;
  const int hlf  = lane >> 4;
  const int r    = lane & 15;
  const int m0b  = blockIdx.y * 32;
  const int n0b  = blockIdx.x * 64;
  const int mrow = (wave >> 2) * 16;            // 0/16 inside the block tile
  const int ncol = (wave & 3) * 16;             // 0/16/32/48

  // staging map: B chunk = 64 rows x 4x16B pieces (256 thr), A = 32 rows x 4 (128 thr)
  const int browi = tid >> 2, bpiece = tid & 3;
  const int arowi = (tid & 127) >> 2, apiece = tid & 3;
  const __bf16* gB = Wt + (size_t)(n0b + browi) * K + bpiece * 8;
  const __bf16* gA = A  + (size_t)(m0b + arowi) * K + apiece * 8;
  __bf16* lB = &sB[browi * LDA + bpiece * 8];
  __bf16* lA = &sA[arowi * LDA + apiece * 8];

  v8f acc = {};
  for (int k0 = 0; k0 < K; k0 += 32) {
    async_copy16(gB + k0, lB);
    if (tid < 128) async_copy16(gA + k0, lA);   // wave-uniform predicate
    async_wait_all();
    __syncthreads();

    const v8bf alo = *(const v8bf*)&sA[(mrow + r) * LDA + hlf * 8];
    const v8bf ahi = *(const v8bf*)&sA[(mrow + r) * LDA + 16 + hlf * 8];
    const v8bf blo = *(const v8bf*)&sB[(ncol + r) * LDA + hlf * 8];
    const v8bf bhi = *(const v8bf*)&sB[(ncol + r) * LDA + 16 + hlf * 8];
    v16bf av, bv;
#pragma unroll
    for (int i = 0; i < 8; ++i) {
      av[i] = alo[i]; av[i + 8] = ahi[i];
      bv[i] = blo[i]; bv[i + 8] = bhi[i];
    }
    acc = __builtin_amdgcn_wmma_f32_16x16x32_bf16(false, av, false, bv,
                                                  (short)0, acc, false, false);
    __syncthreads();                            // LDS reuse fence
  }

  const float bval = bias[n0b + ncol + r];
#pragma unroll
  for (int v = 0; v < 8; ++v) {
    float xv = acc[v] + bval;     // C layout: row m0b+mrow+hlf*8+v, col n0b+ncol+r
    if (ACT == 1) xv = 0.5f * xv * (1.0f + erff(xv * 0.70710678118f)); // exact GELU
    const size_t oi = (size_t)(m0b + mrow + hlf * 8 + v) * N + n0b + ncol + r;
    if (OUTB) Cb[oi] = (__bf16)xv; else Cf[oi] = xv;
  }
}

// ---------------------------------------------------------------------------
// Flash attention with domain masking. One wave per (b, head, 16-query tile).
// qkv layout: [b*S+t][384] with Q=0..127, K=128..255, V=256..383; head h uses
// cols h*16..h*16+15. hd=16 padded to WMMA K=32 with zero fragment halves.
// Output written directly as bf16 (input of the Wo GEMM).
// ---------------------------------------------------------------------------
__global__ __launch_bounds__(32)
void k_attn(const float* __restrict__ qkv, const int* __restrict__ bnd,
            const int* __restrict__ msk, __bf16* __restrict__ obf)
{
  __shared__ __align__(16) __bf16 pt[16 * 16];   // probs tile [q][key]
  __shared__ __align__(16) __bf16 vt[16 * 16];   // V tile transposed [d][key]

  const int bid = blockIdx.x;
  const int qt  = bid & 31;          // S/16 query tiles
  const int hh  = (bid >> 5) & 7;    // head
  const int b   = bid >> 8;          // batch
  const int lane = threadIdx.x;
  const int hlf = lane >> 4;
  const int r   = lane & 15;
  const int q0  = qt * 16;

  // Q fragment (A operand): row = r; real d = hlf*8..+7 in lo; hi = pad(0)
  v16bf qf;
#pragma unroll
  for (int i = 0; i < 16; ++i) qf[i] = (__bf16)0.0f;
  {
    const float* qb = qkv + (size_t)(b * SS + q0 + r) * 384 + hh * 16 + hlf * 8;
#pragma unroll
    for (int i = 0; i < 8; ++i) qf[i] = (__bf16)qb[i];
  }

  int   bq[8];
  float mrun[8], lrun[8];
  v8f oacc = {};
#pragma unroll
  for (int v = 0; v < 8; ++v) {
    const int q = q0 + hlf * 8 + v;
    int bb = bnd[b * SS + q];
    if (!msk[b * SS + q]) bb = 0x40000000;       // invalid q never matches
    bq[v] = bb; mrun[v] = -3.0e30f; lrun[v] = 0.0f;
  }
  const float scale = 0.25f;                     // 1/sqrt(16)

  for (int kt = 0; kt < 32; ++kt) {
    const int key = kt * 16 + r;
    // K^T fragment (B operand): col = key; real d = hlf*8..+7 in lo; hi = pad(0)
    v16bf kf;
#pragma unroll
    for (int i = 0; i < 16; ++i) kf[i] = (__bf16)0.0f;
    {
      const float* kb = qkv + (size_t)(b * SS + key) * 384 + 128 + hh * 16 + hlf * 8;
#pragma unroll
      for (int i = 0; i < 8; ++i) kf[i] = (__bf16)kb[i];
    }
    v8f s = {};
    s = __builtin_amdgcn_wmma_f32_16x16x32_bf16(false, qf, false, kf,
                                                (short)0, s, false, false);
    const int bk = bnd[b * SS + key];
    const int mk = msk[b * SS + key];

    // stage V tile transposed into LDS: vt[d][key]
#pragma unroll
    for (int i = 0; i < 8; ++i) {
      const int idx = lane * 8 + i;
      const int kk = idx >> 4, dd = idx & 15;
      vt[dd * 16 + kk] =
          (__bf16)qkv[(size_t)(b * SS + kt * 16 + kk) * 384 + 256 + hh * 16 + dd];
    }

    // online softmax; C layout: s[v] = score[q0+hlf*8+v][key]
#pragma unroll
    for (int v = 0; v < 8; ++v) {
      const bool allow = (bk == bq[v]) && mk;
      const float sc = allow ? s[v] * scale : -3.0e30f;
      float tm = sc;                              // row-max over the 16 lanes
      tm = fmaxf(tm, __shfl_xor(tm, 1, 32));
      tm = fmaxf(tm, __shfl_xor(tm, 2, 32));
      tm = fmaxf(tm, __shfl_xor(tm, 4, 32));
      tm = fmaxf(tm, __shfl_xor(tm, 8, 32));
      const float mnew = fmaxf(mrun[v], tm);
      const float pv = __expf(sc - mnew);
      float ps = pv;
      ps += __shfl_xor(ps, 1, 32);
      ps += __shfl_xor(ps, 2, 32);
      ps += __shfl_xor(ps, 4, 32);
      ps += __shfl_xor(ps, 8, 32);
      const float corr = __expf(mrun[v] - mnew);
      lrun[v] = lrun[v] * corr + ps;
      mrun[v] = mnew;
      oacc[v] *= corr;
      pt[(hlf * 8 + v) * 16 + r] = (__bf16)pv;    // transpose via LDS
    }
    __syncthreads();

    // P fragment (A): row q = r, keys hlf*8..+7 in lo, hi pad.
    // V fragment (B): col d = r, keys hlf*8..+7 in lo, hi pad.
    v16bf pf, vf;
#pragma unroll
    for (int i = 0; i < 16; ++i) { pf[i] = (__bf16)0.0f; vf[i] = (__bf16)0.0f; }
    {
      const v8bf pl = *(const v8bf*)&pt[r * 16 + hlf * 8];   // ds_load_b128
#pragma unroll
      for (int i = 0; i < 8; ++i) pf[i] = pl[i];
    }
    {
      const v8bf v0 = *(const v8bf*)&vt[r * 16 + hlf * 8];
#pragma unroll
      for (int i = 0; i < 8; ++i) vf[i] = v0[i];
    }
    oacc = __builtin_amdgcn_wmma_f32_16x16x32_bf16(false, pf, false, vf,
                                                   (short)0, oacc, false, false);
    __syncthreads();
  }

#pragma unroll
  for (int v = 0; v < 8; ++v) {
    const int q = q0 + hlf * 8 + v;
    const float o = oacc[v] / fmaxf(lrun[v], 1e-20f);
    obf[(size_t)(b * SS + q) * DD + hh * 16 + r] = (__bf16)o;
  }
}

// ---------------------------------------------------------------------------
// fused residual + LayerNorm, one wave per row of 128
// ---------------------------------------------------------------------------
__global__ __launch_bounds__(32)
void k_ln(float* __restrict__ h, const float* __restrict__ res,
          const float* __restrict__ g, const float* __restrict__ be)
{
  const int row = blockIdx.x;
  const int lane = threadIdx.x;
  float x[4];
  float s = 0.0f;
#pragma unroll
  for (int i = 0; i < 4; ++i) {
    float v = h[(size_t)row * DD + lane * 4 + i];
    if (res) v += res[(size_t)row * DD + lane * 4 + i];
    x[i] = v; s += v;
  }
  s += __shfl_xor(s, 1, 32);  s += __shfl_xor(s, 2, 32);
  s += __shfl_xor(s, 4, 32);  s += __shfl_xor(s, 8, 32);
  s += __shfl_xor(s, 16, 32);
  const float mean = s * (1.0f / 128.0f);
  float vs = 0.0f;
#pragma unroll
  for (int i = 0; i < 4; ++i) { const float d = x[i] - mean; vs += d * d; }
  vs += __shfl_xor(vs, 1, 32);  vs += __shfl_xor(vs, 2, 32);
  vs += __shfl_xor(vs, 4, 32);  vs += __shfl_xor(vs, 8, 32);
  vs += __shfl_xor(vs, 16, 32);
  const float inv = rsqrtf(vs * (1.0f / 128.0f) + 1e-5f);
#pragma unroll
  for (int i = 0; i < 4; ++i) {
    const int c = lane * 4 + i;
    h[(size_t)row * DD + c] = (x[i] - mean) * inv * g[c] + be[c];
  }
}

// ---------------------------------------------------------------------------
// segment-mean pooling
// ---------------------------------------------------------------------------
__global__ void k_zero(float* __restrict__ p, int n)
{
  int i = blockIdx.x * 256 + threadIdx.x;
  if (i < n) p[i] = 0.0f;
}

__global__ __launch_bounds__(128)
void k_pool(const float* __restrict__ h, const int* __restrict__ gid,
            const int* __restrict__ msk, float* __restrict__ sums,
            float* __restrict__ cnts)
{
  const int m = blockIdx.x;
  const int d = threadIdx.x;
  const int id = gid[m];
  const float w = (float)msk[m];
  atomicAdd(&sums[(size_t)id * DD + d], h[(size_t)m * DD + d] * w);
  if (d == 0) atomicAdd(&cnts[id], w);
}

__global__ void k_div(const float* __restrict__ sums, const float* __restrict__ cnts,
                      float* __restrict__ out, int n)
{
  int i = blockIdx.x * 256 + threadIdx.x;
  if (i >= n) return;
  out[i] = sums[i] / fmaxf(cnts[i >> 7], 1.0f);
}

// ---------------------------------------------------------------------------
extern "C" void kernel_launch(void* const* d_in, const int* in_sizes, int n_in,
                              void* d_out, int out_size, void* d_ws, size_t ws_size,
                              hipStream_t stream)
{
  (void)in_sizes; (void)n_in; (void)out_size; (void)ws_size;
  const float* x    = (const float*)d_in[0];
  const int*   bnd  = (const int*)d_in[1];
  const int*   msk  = (const int*)d_in[2];
  const int*   gid  = (const int*)d_in[3];
  // d_in[4] = total_doms (compile-time 2048)
  const float* Wp   = (const float*)d_in[5];
  const float* bp   = (const float*)d_in[6];
  const float* Wqkv = (const float*)d_in[7];
  const float* bqkv = (const float*)d_in[8];
  const float* Wo   = (const float*)d_in[9];
  const float* bo   = (const float*)d_in[10];
  const float* W1   = (const float*)d_in[11];
  const float* b1   = (const float*)d_in[12];
  const float* W2   = (const float*)d_in[13];
  const float* b2   = (const float*)d_in[14];
  const float* g1   = (const float*)d_in[15];
  const float* be1  = (const float*)d_in[16];
  const float* g2   = (const float*)d_in[17];
  const float* be2  = (const float*)d_in[18];
  const float* gf   = (const float*)d_in[19];
  const float* bfin = (const float*)d_in[20];

  char* wsp = (char*)d_ws;
  auto take = [&](size_t bytes) -> void* {
    void* p = (void*)wsp;
    wsp += (bytes + 255) & ~(size_t)255;
    return p;
  };
  float*  h     = (float*)take((size_t)MT * DD * 4);          // activations
  float*  tmp   = (float*)take((size_t)MT * DD * 4);          // D-dim gemm out
  float*  qkv   = (float*)take((size_t)MT * 3 * DD * 4);      // QKV f32
  __bf16* bufA  = (__bf16*)take((size_t)MT * DD * 2);         // bf16 act / attn-out
  __bf16* bufB  = (__bf16*)take((size_t)MT * DFFq * 2);       // bf16 GELU out
  __bf16* wqkvT = (__bf16*)take((size_t)LL * 3 * DD * DD * 2);
  __bf16* woT   = (__bf16*)take((size_t)LL * DD * DD * 2);
  __bf16* w1T   = (__bf16*)take((size_t)LL * DFFq * DD * 2);
  __bf16* w2T   = (__bf16*)take((size_t)LL * DD * DFFq * 2);
  float*  sums  = (float*)take((size_t)TDq * DD * 4 + TDq * 4);
  float*  cnts  = sums + (size_t)TDq * DD;

  // weight prep: transpose to [N][K] + convert to bf16
  for (int l = 0; l < LL; ++l) {
    k_wt<<<(DD * 3 * DD + 255) / 256, 256, 0, stream>>>(
        Wqkv + (size_t)l * DD * 3 * DD, wqkvT + (size_t)l * 3 * DD * DD, DD, 3 * DD);
    k_wt<<<(DD * DD + 255) / 256, 256, 0, stream>>>(
        Wo + (size_t)l * DD * DD, woT + (size_t)l * DD * DD, DD, DD);
    k_wt<<<(DD * DFFq + 255) / 256, 256, 0, stream>>>(
        W1 + (size_t)l * DD * DFFq, w1T + (size_t)l * DFFq * DD, DD, DFFq);
    k_wt<<<(DFFq * DD + 255) / 256, 256, 0, stream>>>(
        W2 + (size_t)l * DFFq * DD, w2T + (size_t)l * DD * DFFq, DFFq, DD);
  }

  k_embed<<<(MT * DD) / 256, 256, 0, stream>>>(x, Wp, bp, h, MT * DD);

  for (int l = 0; l < LL; ++l) {
    k_cvt<<<(MT * DD) / 256, 256, 0, stream>>>(h, bufA, MT * DD);
    k_gemm<0, 0><<<dim3(3 * DD / 64, MT / 32), 256, 0, stream>>>(
        bufA, wqkvT + (size_t)l * 3 * DD * DD, bqkv + l * 3 * DD, qkv, nullptr,
        MT, 3 * DD, DD);
    k_attn<<<BB * NHh * (SS / 16), 32, 0, stream>>>(qkv, bnd, msk, bufA);
    k_gemm<0, 0><<<dim3(DD / 64, MT / 32), 256, 0, stream>>>(
        bufA, woT + (size_t)l * DD * DD, bo + l * DD, tmp, nullptr, MT, DD, DD);
    k_ln<<<MT, 32, 0, stream>>>(h, tmp, g1 + l * DD, be1 + l * DD);
    k_cvt<<<(MT * DD) / 256, 256, 0, stream>>>(h, bufA, MT * DD);
    k_gemm<1, 1><<<dim3(DFFq / 64, MT / 32), 256, 0, stream>>>(
        bufA, w1T + (size_t)l * DFFq * DD, b1 + l * DFFq, nullptr, bufB,
        MT, DFFq, DD);
    k_gemm<0, 0><<<dim3(DD / 64, MT / 32), 256, 0, stream>>>(
        bufB, w2T + (size_t)l * DD * DFFq, b2 + l * DD, tmp, nullptr, MT, DD, DFFq);
    k_ln<<<MT, 32, 0, stream>>>(h, tmp, g2 + l * DD, be2 + l * DD);
  }
  k_ln<<<MT, 32, 0, stream>>>(h, nullptr, gf, bfin);

  k_zero<<<(TDq * DD + TDq + 255) / 256, 256, 0, stream>>>(sums, TDq * DD + TDq);
  k_pool<<<MT, 128, 0, stream>>>(h, gid, msk, sums, cnts);
  k_div<<<(TDq * DD) / 256, 256, 0, stream>>>(sums, cnts, (float*)d_out, TDq * DD);
}
